// OptiXAttentionReal_7464653161177
// MI455X (gfx1250) — matrix operand order repaired and checked by reading, not compiled, not run on hardware
//
#include <hip/hip_runtime.h>
#include <math.h>

// Problem constants (reference): B=2, S=2048, D=1024, H=16, hd=64
#define BB 2
#define SS 2048
#define DD 1024
#define NROWS (BB * SS)          // 4096 token rows
#define ATTN_SCALE 0.125f        // 1/sqrt(64)

typedef float    v4f  __attribute__((ext_vector_type(4)));
typedef float    v8f  __attribute__((ext_vector_type(8)));
typedef _Float16 v4h  __attribute__((ext_vector_type(4)));
typedef _Float16 v8h  __attribute__((ext_vector_type(8)));
typedef _Float16 v16h __attribute__((ext_vector_type(16)));

constexpr int KC  = 64;   // K chunk (2 wmma of K=32 each)
constexpr int AST = 72;   // halves per LDS row (64 + pad; 144B rows, conflict-free)

__device__ inline v16h cat8(v8h lo, v8h hi) {
    return __builtin_shufflevector(lo, hi, 0,1,2,3,4,5,6,7,8,9,10,11,12,13,14,15);
}

// ---- A-chunk staging: 128 rows x 64 k into f16 LDS (row-major, stride AST) ----
__device__ inline void stage_A(const float* __restrict__ A, _Float16* As,
                               int m_base, int k0, int t) {
    #pragma unroll
    for (int i = 0; i < 8; ++i) {
        const int flat4 = t + i * 256;
        const int row = flat4 >> 4, c4 = flat4 & 15;
        v4f v = *(const v4f*)(A + (size_t)(m_base + row) * DD + k0 + c4 * 4);
        v4h h = { (_Float16)v.x, (_Float16)v.y, (_Float16)v.z, (_Float16)v.w };
        *(v4h*)(&As[row * AST + c4 * 4]) = h;
    }
}
__device__ inline void stage_A(const _Float16* __restrict__ A, _Float16* As,
                               int m_base, int k0, int t) {
    #pragma unroll
    for (int i = 0; i < 4; ++i) {
        const int flat8 = t + i * 256;
        const int row = flat8 >> 3, c8 = flat8 & 7;
        v8h v = *(const v8h*)(A + (size_t)(m_base + row) * DD + k0 + c8 * 8);
        *(v8h*)(&As[row * AST + c8 * 8]) = v;
    }
}

__device__ inline void store_c(float* C, size_t i, float v)    { C[i] = v; }
__device__ inline void store_c(_Float16* C, size_t i, float v) { C[i] = (_Float16)v; }

// ---------------------------------------------------------------------------
// GEMM: C[4096,1024] = A @ W^T + bias   (W[n,k] row-major f32 -> f16)
// 256 thr / 8 waves; tile 128M x 16N; v_wmma_f32_16x16x32_f16 (K=32/instr).
// B operand kept n-major in LDS: since b[idx] = W[n=col][k = 32m + 16*hi + idx],
// each lane's fragment is one contiguous 32-byte read (2x ds_load_b128).
// A fragment (16x32): a[idx] = A[row][k = 32m + 16*hi + idx], same k order as B.
// TRANS_OUT=true writes C transposed per batch as [DD][SS] (f16 Vproj^T).
// ---------------------------------------------------------------------------
template <typename TA, typename TO, bool TRANS_OUT>
__global__ __launch_bounds__(256)
void gemm_f16wmma(const TA* __restrict__ A, const float* __restrict__ W,
                  const float* __restrict__ bias, TO* __restrict__ C) {
    __shared__ _Float16 As[128 * AST];   // 18 KB
    __shared__ _Float16 Bs[16 * AST];    // 2.25 KB: row n (0..15), halves along k

    const int t    = threadIdx.x;
    const int wave = t >> 5;
    const int lane = t & 31;
    const int col  = lane & 15;
    const int hi   = lane >> 4;
    const int m_base = blockIdx.x * 128;
    const int n0     = blockIdx.y * 16;

    v8f c = {0.f, 0.f, 0.f, 0.f, 0.f, 0.f, 0.f, 0.f};

    for (int k0 = 0; k0 < DD; k0 += KC) {
        stage_A(A, As, m_base, k0, t);
        // Stage W tile (16n x 64k) n-major: 1 v4f load + v4h store per thread.
        {
            const int n = t >> 4, c4 = t & 15;
            v4f v = *(const v4f*)(W + (size_t)(n0 + n) * DD + k0 + c4 * 4);
            v4h h = { (_Float16)v.x, (_Float16)v.y, (_Float16)v.z, (_Float16)v.w };
            *(v4h*)(&Bs[n * AST + c4 * 4]) = h;
        }
        __syncthreads();

        if (k0 + KC < DD)
            __builtin_prefetch(A + (size_t)(m_base + (t >> 1)) * DD + (k0 + KC) + (t & 1) * 32, 0, 1);

        const _Float16* ap = &As[(wave * 16 + col) * AST];
        const _Float16* bp = &Bs[col * AST];
        #pragma unroll
        for (int m = 0; m < 2; ++m) {
            v8h alo = *(const v8h*)(ap + 32 * m + 16 * hi);
            v8h ahi = *(const v8h*)(ap + 32 * m + 16 * hi + 8);
            v16h a  = cat8(alo, ahi);
            v16h b  = *(const v16h*)(bp + 32 * m + 16 * hi);
            c = __builtin_amdgcn_wmma_f32_16x16x32_f16(false, a, false, b, (short)0, c, false, false);
        }
        __syncthreads();
    }

    const float bval = bias[n0 + col];
    if constexpr (TRANS_OUT) {
        // C layout: [BB][DD][SS]; 128-row tiles never straddle batches.
        const int bb = m_base / SS;
        const int m0 = (m_base % SS) + wave * 16 + hi * 8;
        #pragma unroll
        for (int r = 0; r < 8; ++r)
            store_c(C, ((size_t)bb * DD + n0 + col) * SS + m0 + r, c[r] + bval);
    } else {
        const int m0 = m_base + wave * 16 + hi * 8;
        #pragma unroll
        for (int r = 0; r < 8; ++r)
            store_c(C, (size_t)(m0 + r) * DD + n0 + col, c[r] + bval);
    }
}

// ---------------------------------------------------------------------------
// pos[row,j] = sum_d x[row,d]*w3d[j,d] + b3d[j]  (fp32, one wave per row)
// ---------------------------------------------------------------------------
__global__ __launch_bounds__(256)
void pos_kernel(const float* __restrict__ x, const float* __restrict__ w3d,
                const float* __restrict__ b3d, float* __restrict__ pos) {
    const int row  = (blockIdx.x * blockDim.x + threadIdx.x) >> 5;
    const int lane = threadIdx.x & 31;
    if (row >= NROWS) return;
    const float* xr = x + (size_t)row * DD;
    float s0 = 0.f, s1 = 0.f, s2 = 0.f;
    for (int d = lane; d < DD; d += 32) {
        const float xv = xr[d];
        s0 += xv * w3d[d];
        s1 += xv * w3d[DD + d];
        s2 += xv * w3d[2 * DD + d];
    }
    #pragma unroll
    for (int off = 16; off; off >>= 1) {
        s0 += __shfl_xor(s0, off, 32);
        s1 += __shfl_xor(s1, off, 32);
        s2 += __shfl_xor(s2, off, 32);
    }
    if (lane == 0) {
        pos[row * 3 + 0] = s0 + b3d[0];
        pos[row * 3 + 1] = s1 + b3d[1];
        pos[row * 3 + 2] = s2 + b3d[2];
    }
}

// ---------------------------------------------------------------------------
// Row max is exactly 0 (diagonal distance = 0, all scores <= 0), so
// invl[q] = 1 / sum_{k<=q} exp(-dist*scale).  fp32, one wave per q.
// ---------------------------------------------------------------------------
__global__ __launch_bounds__(256)
void denom_kernel(const float* __restrict__ pos, float* __restrict__ invl) {
    const int wg   = (blockIdx.x * blockDim.x + threadIdx.x) >> 5;
    const int lane = threadIdx.x & 31;
    if (wg >= NROWS) return;
    const int bb = wg / SS, q = wg % SS;
    const float* pb = pos + (size_t)bb * SS * 3;
    const float qx = pb[q * 3], qy = pb[q * 3 + 1], qz = pb[q * 3 + 2];
    float s = 0.f;
    for (int k = lane; k <= q; k += 32) {
        const float dx = qx - pb[k * 3];
        const float dy = qy - pb[k * 3 + 1];
        const float dz = qz - pb[k * 3 + 2];
        s += __expf(-__builtin_amdgcn_sqrtf(dx * dx + dy * dy + dz * dz) * ATTN_SCALE);
    }
    #pragma unroll
    for (int off = 16; off; off >>= 1) s += __shfl_xor(s, off, 32);
    if (lane == 0) invl[wg] = 1.f / s;   // s >= 1 always
}

// ---------------------------------------------------------------------------
// ctx[b,q,:] = sum_{k<=q} P[q,k] * V[b,k,:],  P = exp(-dist*scale)/l_q.
// V supplied transposed per batch as Vt[b][d][s] (f16) so each lane's
// B fragment is 16 contiguous K halves of one column d: no LDS swizzle.
// Grid (NROWS/16, DD/128); wave w owns 16 d-columns.
// ---------------------------------------------------------------------------
__global__ __launch_bounds__(256)
void ctx_kernel(const float* __restrict__ pos, const float* __restrict__ invl,
                const _Float16* __restrict__ Vt, _Float16* __restrict__ ctx) {
    __shared__ _Float16 Ps[16 * AST];    // 2.25 KB (P tile, A-layout rows)
    __shared__ _Float16 Vls[128 * AST];  // 18 KB: row = d (0..127), halves along k

    const int t    = threadIdx.x;
    const int wave = t >> 5;
    const int lane = t & 31;
    const int col  = lane & 15;
    const int hi   = lane >> 4;
    const int qg0  = blockIdx.x * 16;    // tiles never straddle batches
    const int bb   = qg0 / SS;
    const int q0   = qg0 % SS;
    const int nb   = blockIdx.y * 128;
    const int n0   = nb + wave * 16;
    const float* pb = pos + (size_t)bb * SS * 3;
    const float* il = invl + (size_t)bb * SS;
    const _Float16* vtb = Vt + (size_t)bb * DD * SS;

    v8f c = {0.f, 0.f, 0.f, 0.f, 0.f, 0.f, 0.f, 0.f};

    const int kend = q0 + 15;
    for (int k0 = 0; k0 <= kend; k0 += KC) {
        // P tile (16q x 64k): fp32 distance/exp, f16 store into row-major A-layout.
        #pragma unroll
        for (int i = 0; i < 4; ++i) {
            const int flat = t + i * 256;
            const int r = flat >> 6, kk = flat & 63;
            const int q = q0 + r, k = k0 + kk;
            float p = 0.f;
            if (k <= q) {
                const float dx = pb[q * 3]     - pb[k * 3];
                const float dy = pb[q * 3 + 1] - pb[k * 3 + 1];
                const float dz = pb[q * 3 + 2] - pb[k * 3 + 2];
                p = __expf(-__builtin_amdgcn_sqrtf(dx * dx + dy * dy + dz * dz) * ATTN_SCALE) * il[q];
            }
            Ps[r * AST + kk] = (_Float16)p;
        }
        // V^T chunk (128 d-rows x 64 k halves): coalesced v8h load + v8h store.
        #pragma unroll
        for (int i = 0; i < 4; ++i) {
            const int flat8 = t + i * 256;
            const int row = flat8 >> 3, c8 = flat8 & 7;
            v8h v = *(const v8h*)(vtb + (size_t)(nb + row) * SS + k0 + c8 * 8);
            *(v8h*)(&Vls[row * AST + c8 * 8]) = v;
        }
        __syncthreads();

        const _Float16* ap = &Ps[col * AST];
        const _Float16* bp = &Vls[(wave * 16 + col) * AST];
        #pragma unroll
        for (int m = 0; m < 2; ++m) {
            v8h alo = *(const v8h*)(ap + 32 * m + 16 * hi);
            v8h ahi = *(const v8h*)(ap + 32 * m + 16 * hi + 8);
            v16h a  = cat8(alo, ahi);
            v16h b  = *(const v16h*)(bp + 32 * m + 16 * hi);
            c = __builtin_amdgcn_wmma_f32_16x16x32_f16(false, a, false, b, (short)0, c, false, false);
        }
        __syncthreads();
    }

    const int qout = q0 + hi * 8;
    #pragma unroll
    for (int r = 0; r < 8; ++r)
        ctx[((size_t)bb * SS + qout + r) * DD + n0 + col] = (_Float16)c[r];
}

// ---------------------------------------------------------------------------
// Launcher. Inputs: x, wq, bq, wk, bk, wv, bv, wo, bo, w3d, b3d.
// Q/K projections are dead code in the reference math.
// ---------------------------------------------------------------------------
extern "C" void kernel_launch(void* const* d_in, const int* in_sizes, int n_in,
                              void* d_out, int out_size, void* d_ws, size_t ws_size,
                              hipStream_t stream) {
    (void)in_sizes; (void)n_in; (void)out_size; (void)ws_size;
    const float* x   = (const float*)d_in[0];
    const float* wv  = (const float*)d_in[5];
    const float* bv  = (const float*)d_in[6];
    const float* wo  = (const float*)d_in[7];
    const float* bo  = (const float*)d_in[8];
    const float* w3d = (const float*)d_in[9];
    const float* b3d = (const float*)d_in[10];
    float* out = (float*)d_out;

    // Workspace: Vproj^T f16 [BB][DD][SS] | ctx f16 [NROWS][DD] | pos f32 | invl f32
    _Float16* Vt   = (_Float16*)d_ws;
    _Float16* ctxb = Vt + (size_t)NROWS * DD;
    float*    pos  = (float*)(ctxb + (size_t)NROWS * DD);
    float*    invl = pos + NROWS * 3;

    const dim3 blk(256);
    const dim3 gemm_grid(NROWS / 128, DD / 16);     // (32, 64)
    const int  rowwave_blocks = (NROWS * 32) / 256; // one wave per row

    gemm_f16wmma<float, _Float16, true><<<gemm_grid, blk, 0, stream>>>(x, wv, bv, Vt);
    pos_kernel<<<rowwave_blocks, blk, 0, stream>>>(x, w3d, b3d, pos);
    denom_kernel<<<rowwave_blocks, blk, 0, stream>>>(pos, invl);
    const dim3 ctx_grid(NROWS / 16, DD / 128);      // (256, 8)
    ctx_kernel<<<ctx_grid, blk, 0, stream>>>(pos, invl, Vt, ctxb);
    gemm_f16wmma<_Float16, float, false><<<gemm_grid, blk, 0, stream>>>(ctxb, wo, bo, out);
}
